// DenoisingEdgeNetwork_25202868093634
// MI455X (gfx1250) — matrix operand order repaired and checked by reading, not compiled, not run on hardware
//
#include <hip/hip_runtime.h>
#include <hip/hip_bf16.h>

typedef _Float16 f16_t;
typedef __attribute__((ext_vector_type(16))) _Float16 v16h;
typedef __attribute__((ext_vector_type(8)))  float    v8f;

#define NN   6144      // nodes
#define EE   288768    // edges
#define BB   128       // graphs
#define NPG  48        // nodes per graph  (48 = 3 tiles of 16)
#define EPG  2256      // edges per graph  (2256 = 141 tiles of 16)
#define SD   256
#define TPB_E 8        // edge tiles per block (18048 / 8 = 2256 blocks)

#ifndef __has_builtin
#define __has_builtin(x) 0
#endif
#if __has_builtin(__builtin_amdgcn_global_load_async_to_lds_b128)
#define HAVE_ASYNC_LDS 1
typedef int i128_t __attribute__((vector_size(16)));           // matches builtin pointee
typedef __attribute__((address_space(1))) i128_t as1_i128;     // global
typedef __attribute__((address_space(3))) i128_t as3_i128;     // LDS
#else
#define HAVE_ASYNC_LDS 0
#endif

__device__ __forceinline__ v8f wmma16(v16h a, v16h b, v8f c) {
  // D = A(16x32 f16) * B(32x16 f16) + C(16x16 f32)
  return __builtin_amdgcn_wmma_f32_16x16x32_f16(false, a, false, b, (short)0, c, false, false);
}
__device__ __forceinline__ v8f splat8(float v) {
  v8f r; for (int i = 0; i < 8; ++i) r[i] = v; return r;
}
__device__ __forceinline__ float silu(float v) {
  return v / (1.0f + __expf(-v));
}
// A-fragment coordinates for element (row m, k_local in [0,32)):
//   lane = m + ((k_local>>3)&1)*16 ; elem = (k_local>>4)*8 + (k_local&7)
__device__ __forceinline__ int afrag_lane(int m, int kl) { return m + (((kl >> 3) & 1) << 4); }
__device__ __forceinline__ int afrag_elem(int kl)        { return ((kl >> 4) << 3) | (kl & 7); }

// 16-byte global -> LDS copy; async path when the toolchain exposes it.
// AS1 ptr == flat bits for global; AS3 ptr == low 32 bits of flat LDS addr
// (ISA 10.2 aperture mapping), so integer casts build valid AS pointers.
__device__ __forceinline__ void copy16_g2l(const void* g, void* l) {
#if HAVE_ASYNC_LDS
  __builtin_amdgcn_global_load_async_to_lds_b128(
      (as1_i128*)(uintptr_t)g,
      (as3_i128*)(unsigned int)(uintptr_t)l, 0, 0);
#else
  *(int4*)l = *(const int4*)g;
#endif
}
__device__ __forceinline__ void copy_wait() {
#if HAVE_ASYNC_LDS
#if __has_builtin(__builtin_amdgcn_s_wait_asynccnt)
  __builtin_amdgcn_s_wait_asynccnt(0);
#else
  asm volatile("s_wait_asynccnt 0" ::: "memory");
#endif
#endif
}
__device__ __forceinline__ void stage_g2l(f16_t* dst, const f16_t* src, int n16,
                                          int tid, int nthr) {
  for (int i = tid; i < n16; i += nthr) copy16_g2l(src + i * 8, dst + i * 8);
}

// ---------------------------------------------------------------------------
// Weight prep: f32 [K,Nout] row-major -> f16 B fragments.
// Fragment (ct,kb): lane l holds column ct*16+(l&15), K = kb*32+(e>>3)*16+(l>>4)*8+(e&7).
// ---------------------------------------------------------------------------
__global__ __launch_bounds__(32) void wprep_kernel(const float* __restrict__ W,
                                                   f16_t* __restrict__ out,
                                                   int K, int Nout) {
  int ct = blockIdx.x, kb = blockIdx.y, nkb = gridDim.y;
  int lane = threadIdx.x;
  int col = ct * 16 + (lane & 15);
  int kbase = kb * 32 + (lane >> 4) * 8;
  f16_t* dst = out + (((size_t)(ct * nkb + kb) * 32 + lane) * 16);
  for (int e = 0; e < 16; ++e) {
    int k = kbase + ((e >> 3) << 4) + (e & 7);
    float v = (k < K && col < Nout) ? W[(size_t)k * Nout + col] : 0.0f;
    dst[e] = (f16_t)v;
  }
}

// ---------------------------------------------------------------------------
// Per-graph prep: centered coords (== coords_pred) + time embeddings ta/tb.
// ---------------------------------------------------------------------------
__global__ __launch_bounds__(64) void graph_prep_kernel(
    const float* __restrict__ t, const float* __restrict__ pos,
    const float* __restrict__ Wta, const float* __restrict__ bta,
    const float* __restrict__ Wtb, const float* __restrict__ btb,
    float* __restrict__ out_coords, float* __restrict__ ta_ws,
    float* __restrict__ tb_ws) {
  __shared__ float lp[NPG][3];
  __shared__ float mean[3];
  int g = blockIdx.x, tid = threadIdx.x;
  if (tid < NPG)
    for (int c = 0; c < 3; ++c) lp[tid][c] = pos[((size_t)g * NPG + tid) * 3 + c];
  __syncthreads();
  if (tid < 3) {
    float s = 0.0f;
    for (int n = 0; n < NPG; ++n) s += lp[n][tid];
    mean[tid] = s * (1.0f / NPG);
  }
  __syncthreads();
  if (tid < NPG)
    for (int c = 0; c < 3; ++c)
      out_coords[((size_t)g * NPG + tid) * 3 + c] = lp[tid][c] - mean[c];
  float tv = t[g];
  for (int k = tid; k < SD; k += 64) ta_ws[g * SD + k] = tv * Wta[k] + bta[k];
  if (tid < 32) tb_ws[g * 32 + tid] = tv * Wtb[tid] + btb[tid];
}

// ---------------------------------------------------------------------------
// Node pipeline: 16 nodes per block, 8 waves, dual-accumulator K loops.
// ---------------------------------------------------------------------------
__global__ __launch_bounds__(256) void node_kernel(
    const float* __restrict__ x, const float* __restrict__ Wam,
    const float* __restrict__ bam, const float* __restrict__ ta_ws,
    const f16_t* __restrict__ fWatm, const float* __restrict__ batm,
    const f16_t* __restrict__ fWsh, const float* __restrict__ bsh,
    const f16_t* __restrict__ fWat, const float* __restrict__ bat,
    f16_t* __restrict__ sbuf, float* __restrict__ out_atoms) {
  __shared__ float lx[16][16];
  __shared__ float lwam[16][SD];
  __shared__ __attribute__((aligned(32))) f16_t A0[8][32][16];
  __shared__ __attribute__((aligned(32))) f16_t A1[8][32][16];
  __shared__ __attribute__((aligned(32))) f16_t A2[8][32][16];

  int tid = threadIdx.x;
  int node0 = blockIdx.x * 16;
  int g = node0 / NPG;  // tiles never straddle graphs (48 = 3*16)

  { // stage x tile and Wam
    int m = tid >> 4, k = tid & 15;
    lx[m][k] = x[(size_t)(node0 + m) * 16 + k];
    for (int o = 0; o < 16; ++o) {
      int idx = tid + o * 256;  // 0..4095 over [16,256]
      lwam[idx >> 8][idx & 255] = Wam[idx];
    }
  }
  __syncthreads();

  { // s0 = x@Wam + bam + ta  -> A0 fragments (f16)
    int m = tid >> 4;
    int nb = (tid & 15) * 16;
    for (int o = 0; o < 16; ++o) {
      int n = nb + o;
      float acc = bam[n] + ta_ws[g * SD + n];
      for (int k = 0; k < 16; ++k) acc += lx[m][k] * lwam[k][n];
      int kb = n >> 5, kl = n & 31;
      A0[kb][afrag_lane(m, kl)][afrag_elem(kl)] = (f16_t)acc;
    }
  }
  __syncthreads();

  int wid = tid >> 5, lane = tid & 31;
  int mrow = (lane >> 4) << 3;  // D row base: m = r + mrow
  int ncol = lane & 15;
  int ct0 = wid, ct1 = wid + 8;
  int n0 = ct0 * 16 + ncol, n1 = ct1 * 16 + ncol;

  // GEMM1: s1 = s0 @ Watm + batm -> A1 (shared A fragment, two B streams)
  {
    v8f acc0 = splat8(batm[n0]);
    v8f acc1 = splat8(batm[n1]);
    for (int kb = 0; kb < 8; ++kb) {
      v16h a = *(const v16h*)(&A0[kb][lane][0]);
      v16h b0 = *(const v16h*)(fWatm + ((size_t)(ct0 * 8 + kb) * 32 + lane) * 16);
      v16h b1 = *(const v16h*)(fWatm + ((size_t)(ct1 * 8 + kb) * 32 + lane) * 16);
      acc0 = wmma16(a, b0, acc0);
      acc1 = wmma16(a, b1, acc1);
    }
    int kb0 = n0 >> 5, al0 = afrag_lane(0, n0 & 31), eo0 = afrag_elem(n0 & 31);
    int kb1 = n1 >> 5, al1 = afrag_lane(0, n1 & 31), eo1 = afrag_elem(n1 & 31);
    for (int r = 0; r < 8; ++r) {
      A1[kb0][(r + mrow) + al0][eo0] = (f16_t)acc0[r];
      A1[kb1][(r + mrow) + al1][eo1] = (f16_t)acc1[r];
    }
  }
  __syncthreads();

  // GEMM2: s2 = silu(s1 @ Wsh + bsh) -> sbuf + A2
  {
    v8f acc0 = splat8(bsh[n0]);
    v8f acc1 = splat8(bsh[n1]);
    for (int kb = 0; kb < 8; ++kb) {
      v16h a = *(const v16h*)(&A1[kb][lane][0]);
      v16h b0 = *(const v16h*)(fWsh + ((size_t)(ct0 * 8 + kb) * 32 + lane) * 16);
      v16h b1 = *(const v16h*)(fWsh + ((size_t)(ct1 * 8 + kb) * 32 + lane) * 16);
      acc0 = wmma16(a, b0, acc0);
      acc1 = wmma16(a, b1, acc1);
    }
    int kb0 = n0 >> 5, al0 = afrag_lane(0, n0 & 31), eo0 = afrag_elem(n0 & 31);
    int kb1 = n1 >> 5, al1 = afrag_lane(0, n1 & 31), eo1 = afrag_elem(n1 & 31);
    for (int r = 0; r < 8; ++r) {
      int m = r + mrow;
      f16_t h0 = (f16_t)silu(acc0[r]);
      f16_t h1 = (f16_t)silu(acc1[r]);
      sbuf[(size_t)(node0 + m) * SD + n0] = h0;
      sbuf[(size_t)(node0 + m) * SD + n1] = h1;
      A2[kb0][m + al0][eo0] = h0;
      A2[kb1][m + al1][eo1] = h1;
    }
  }
  __syncthreads();

  // atoms = s2 @ Wat + bat  (Nout = 16, wave 0)
  if (wid == 0) {
    v8f acc = splat8(bat[ncol]);
    for (int kb = 0; kb < 8; ++kb) {
      v16h a = *(const v16h*)(&A2[kb][lane][0]);
      v16h b = *(const v16h*)(fWat + ((size_t)kb * 32 + lane) * 16);
      acc = wmma16(a, b, acc);
    }
    for (int r = 0; r < 8; ++r)
      out_atoms[(size_t)(node0 + r + mrow) * 16 + ncol] = acc[r];
  }
}

// ---------------------------------------------------------------------------
// Edge pipeline: LDS-resident weights, TPB_E tiles of 16 edges per block.
// ---------------------------------------------------------------------------
__global__ __launch_bounds__(256) void edge_kernel(
    const float* __restrict__ eattr, const int* __restrict__ eidx,
    const float* __restrict__ coords, const f16_t* __restrict__ sbuf,
    const float* __restrict__ tb_ws, const float* __restrict__ Wbm,
    const float* __restrict__ bbm, const f16_t* __restrict__ fWbtm,
    const float* __restrict__ bbtm, const f16_t* __restrict__ fWhbm,
    const float* __restrict__ bhbm, const f16_t* __restrict__ fW0,
    const float* __restrict__ W0, const float* __restrict__ b0,
    const f16_t* __restrict__ fW1, const float* __restrict__ b1,
    float* __restrict__ out_bonds) {
  // LDS-resident weight fragments (CDNA5: up to 320KB per workgroup)
  __shared__ __attribute__((aligned(32))) f16_t LW0[16 * 8 * 512];   // 128KB
  __shared__ __attribute__((aligned(32))) f16_t LWhbm[16 * 512];     // 16KB
  __shared__ __attribute__((aligned(32))) f16_t LWbtm[2 * 512];      // 2KB
  __shared__ __attribute__((aligned(32))) f16_t LW1[8 * 512];        // 8KB
  // per-tile working set
  __shared__ float ssum[16][SD];
  __shared__ float lea[16][5];
  __shared__ float dl[16];
  __shared__ int li[16], lj[16];
  __shared__ __attribute__((aligned(32))) f16_t AE[32][16];
  __shared__ __attribute__((aligned(32))) f16_t AE2[32][16];
  __shared__ __attribute__((aligned(32))) f16_t AF[8][32][16];
  __shared__ __attribute__((aligned(32))) f16_t AG[8][32][16];

  int tid = threadIdx.x;
  int wid = tid >> 5, lane = tid & 31;
  int mrow = (lane >> 4) << 3;
  int ncol = lane & 15;
  int ct0 = wid, ct1 = wid + 8;
  int n0 = ct0 * 16 + ncol, n1 = ct1 * 16 + ncol;

  // stage all edge-stage weights into LDS once per block (async when available)
  stage_g2l(LW0, fW0, 16 * 8 * 512 / 8, tid, 256);
  stage_g2l(LWhbm, fWhbm, 16 * 512 / 8, tid, 256);
  stage_g2l(LWbtm, fWbtm, 2 * 512 / 8, tid, 256);
  stage_g2l(LW1, fW1, 8 * 512 / 8, tid, 256);
  copy_wait();
  __syncthreads();

  for (int tt = 0; tt < TPB_E; ++tt) {
    int tile = blockIdx.x * TPB_E + tt;
    int e0i = tile * 16;
    int g = e0i / EPG;  // tiles never straddle graphs (2256 = 141*16)

    if (tid < 16) {
      int e = e0i + tid;
      int jj = eidx[e];       // edge_index[0] = j (src)
      int ii = eidx[EE + e];  // edge_index[1] = i (dst)
      li[tid] = ii; lj[tid] = jj;
      float dx = coords[(size_t)ii * 3 + 0] - coords[(size_t)jj * 3 + 0];
      float dy = coords[(size_t)ii * 3 + 1] - coords[(size_t)jj * 3 + 1];
      float dz = coords[(size_t)ii * 3 + 2] - coords[(size_t)jj * 3 + 2];
      dl[tid] = sqrtf(dx * dx + dy * dy + dz * dz);
      for (int k = 0; k < 5; ++k) lea[tid][k] = eattr[(size_t)e * 5 + k];
    }
    __syncthreads();

    { // gather s[i] + s[j]
      int m = tid >> 4;
      int nb = (tid & 15) * 16;
      const f16_t* si = sbuf + (size_t)li[m] * SD;
      const f16_t* sj = sbuf + (size_t)lj[m] * SD;
      for (int o = 0; o < 16; ++o) {
        int n = nb + o;
        ssum[m][n] = (float)si[n] + (float)sj[n];
      }
    }
    { // e0 = ea@Wbm + bbm + tb  (16x32, K=5 VALU) -> AE fragments
      int m = tid >> 4;
      int c0 = (tid & 15) * 2;
      for (int o = 0; o < 2; ++o) {
        int c = c0 + o;
        float acc = bbm[c] + tb_ws[g * 32 + c];
        for (int k = 0; k < 5; ++k) acc += lea[m][k] * Wbm[k * 32 + c];
        AE[afrag_lane(m, c)][afrag_elem(c)] = (f16_t)acc;
      }
    }
    __syncthreads();

    // e1 = e0 @ Wbtm + bbtm  (2 col-tiles, waves 0..1) -> AE2
    if (wid < 2) {
      int n = wid * 16 + ncol;
      v8f acc = splat8(bbtm[n]);
      v16h a = *(const v16h*)(&AE[lane][0]);
      v16h b = *(const v16h*)(&LWbtm[((size_t)wid * 32 + lane) * 16]);
      acc = wmma16(a, b, acc);
      int al = afrag_lane(0, n), eo = afrag_elem(n);
      for (int r = 0; r < 8; ++r) AE2[(r + mrow) + al][eo] = (f16_t)acc[r];
    }
    __syncthreads();

    // f = e1 @ Whbm + bhbm + (s[i]+s[j])  -> AF
    {
      v8f acc0 = splat8(bhbm[n0]);
      v8f acc1 = splat8(bhbm[n1]);
      v16h a = *(const v16h*)(&AE2[lane][0]);
      v16h b0 = *(const v16h*)(&LWhbm[((size_t)ct0 * 32 + lane) * 16]);
      v16h b1 = *(const v16h*)(&LWhbm[((size_t)ct1 * 32 + lane) * 16]);
      acc0 = wmma16(a, b0, acc0);
      acc1 = wmma16(a, b1, acc1);
      int kb0 = n0 >> 5, al0 = afrag_lane(0, n0 & 31), eo0 = afrag_elem(n0 & 31);
      int kb1 = n1 >> 5, al1 = afrag_lane(0, n1 & 31), eo1 = afrag_elem(n1 & 31);
      for (int r = 0; r < 8; ++r) {
        int m = r + mrow;
        AF[kb0][m + al0][eo0] = (f16_t)(acc0[r] + ssum[m][n0]);
        AF[kb1][m + al1][eo1] = (f16_t)(acc1[r] + ssum[m][n1]);
      }
    }
    __syncthreads();

    // g = silu([f,d] @ W0 + b0): K=256 WMMA + rank-1 d*W0[256,:] in acc init
    {
      float bn0 = b0[n0], wl0 = W0[(size_t)256 * SD + n0];
      float bn1 = b0[n1], wl1 = W0[(size_t)256 * SD + n1];
      v8f acc0, acc1;
      for (int r = 0; r < 8; ++r) {
        float dr = dl[r + mrow];
        acc0[r] = bn0 + dr * wl0;
        acc1[r] = bn1 + dr * wl1;
      }
      for (int kb = 0; kb < 8; ++kb) {
        v16h a = *(const v16h*)(&AF[kb][lane][0]);
        v16h b0v = *(const v16h*)(&LW0[((size_t)(ct0 * 8 + kb) * 32 + lane) * 16]);
        v16h b1v = *(const v16h*)(&LW0[((size_t)(ct1 * 8 + kb) * 32 + lane) * 16]);
        acc0 = wmma16(a, b0v, acc0);
        acc1 = wmma16(a, b1v, acc1);
      }
      int kb0 = n0 >> 5, al0 = afrag_lane(0, n0 & 31), eo0 = afrag_elem(n0 & 31);
      int kb1 = n1 >> 5, al1 = afrag_lane(0, n1 & 31), eo1 = afrag_elem(n1 & 31);
      for (int r = 0; r < 8; ++r) {
        AG[kb0][(r + mrow) + al0][eo0] = (f16_t)silu(acc0[r]);
        AG[kb1][(r + mrow) + al1][eo1] = (f16_t)silu(acc1[r]);
      }
    }
    __syncthreads();

    // bonds = g @ W1 + b1  (Nout=5 padded to 16, wave 0)
    if (wid == 0) {
      v8f acc = splat8((ncol < 5) ? b1[ncol] : 0.0f);
      for (int kb = 0; kb < 8; ++kb) {
        v16h a = *(const v16h*)(&AG[kb][lane][0]);
        v16h b = *(const v16h*)(&LW1[((size_t)kb * 32 + lane) * 16]);
        acc = wmma16(a, b, acc);
      }
      if (ncol < 5)
        for (int r = 0; r < 8; ++r)
          out_bonds[(size_t)(e0i + r + mrow) * 5 + ncol] = acc[r];
    }
    // next-tile writes are fenced by the barrier after the tid<16 stage
  }
}

// ---------------------------------------------------------------------------
extern "C" void kernel_launch(void* const* d_in, const int* in_sizes, int n_in,
                              void* d_out, int out_size, void* d_ws,
                              size_t ws_size, hipStream_t stream) {
  const float* x    = (const float*)d_in[0];
  const float* t    = (const float*)d_in[1];
  const float* pos  = (const float*)d_in[2];
  const float* ea   = (const float*)d_in[3];
  const int*   eidx = (const int*)d_in[4];
  // d_in[5] batch, d_in[6] batch_edge: computed arithmetically (contiguous layout)
  const float* Wta  = (const float*)d_in[7];  const float* bta  = (const float*)d_in[8];
  const float* Wtb  = (const float*)d_in[9];  const float* btb  = (const float*)d_in[10];
  const float* Wam  = (const float*)d_in[11]; const float* bam  = (const float*)d_in[12];
  const float* Watm = (const float*)d_in[13]; const float* batm = (const float*)d_in[14];
  const float* Wbm  = (const float*)d_in[15]; const float* bbm  = (const float*)d_in[16];
  const float* Wbtm = (const float*)d_in[17]; const float* bbtm = (const float*)d_in[18];
  const float* Wsh  = (const float*)d_in[19]; const float* bsh  = (const float*)d_in[20];
  const float* Whbm = (const float*)d_in[21]; const float* bhbm = (const float*)d_in[22];
  const float* W0   = (const float*)d_in[23]; const float* b0   = (const float*)d_in[24];
  const float* W1   = (const float*)d_in[25]; const float* b1   = (const float*)d_in[26];
  // d_in[27] coords weight unused (v == 0)
  const float* Wat  = (const float*)d_in[28]; const float* bat  = (const float*)d_in[29];

  float* out        = (float*)d_out;
  float* out_coords = out;                        // [N,3]
  float* out_atoms  = out + (size_t)NN * 3;       // [N,16]
  float* out_bonds  = out + (size_t)NN * 3 + (size_t)NN * 16;  // [E,5]

  // workspace layout (all offsets well aligned)
  char* ws = (char*)d_ws;
  size_t off = 0;
  float* ta_ws = (float*)(ws + off); off += (size_t)BB * SD * 4;   // 128K
  float* tb_ws = (float*)(ws + off); off += (size_t)BB * 32 * 4;   // 16K
  f16_t* sbuf  = (f16_t*)(ws + off); off += (size_t)NN * SD * 2;   // 3M
  f16_t* fWatm = (f16_t*)(ws + off); off += 16 * 8 * 1024;
  f16_t* fWsh  = (f16_t*)(ws + off); off += 16 * 8 * 1024;
  f16_t* fWat  = (f16_t*)(ws + off); off += 1 * 8 * 1024;
  f16_t* fWhbm = (f16_t*)(ws + off); off += 16 * 1 * 1024;
  f16_t* fWbtm = (f16_t*)(ws + off); off += 2 * 1 * 1024;
  f16_t* fW0   = (f16_t*)(ws + off); off += 16 * 8 * 1024;
  f16_t* fW1   = (f16_t*)(ws + off); off += 1 * 8 * 1024;
  (void)off; (void)ws_size; (void)in_sizes; (void)n_in; (void)out_size;

  // 1) weight fragment prep
  wprep_kernel<<<dim3(16, 8), 32, 0, stream>>>(Watm, fWatm, 256, 256);
  wprep_kernel<<<dim3(16, 8), 32, 0, stream>>>(Wsh,  fWsh,  256, 256);
  wprep_kernel<<<dim3(1, 8),  32, 0, stream>>>(Wat,  fWat,  256, 16);
  wprep_kernel<<<dim3(16, 1), 32, 0, stream>>>(Whbm, fWhbm, 32,  256);
  wprep_kernel<<<dim3(2, 1),  32, 0, stream>>>(Wbtm, fWbtm, 32,  32);
  wprep_kernel<<<dim3(16, 8), 32, 0, stream>>>(W0,   fW0,   256, 256);  // rows 0..255
  wprep_kernel<<<dim3(1, 8),  32, 0, stream>>>(W1,   fW1,   256, 5);

  // 2) per-graph centering + time embeddings (writes coords_pred)
  graph_prep_kernel<<<BB, 64, 0, stream>>>(t, pos, Wta, bta, Wtb, btb,
                                           out_coords, ta_ws, tb_ws);

  // 3) node pipeline (writes atoms_pred, sbuf)
  node_kernel<<<NN / 16, 256, 0, stream>>>(x, Wam, bam, ta_ws, fWatm, batm,
                                           fWsh, bsh, fWat, bat, sbuf,
                                           out_atoms);

  // 4) edge pipeline (writes bonds_pred)
  edge_kernel<<<EE / 16 / TPB_E, 256, 0, stream>>>(
      ea, eidx, out_coords, sbuf, tb_ws, Wbm, bbm, fWbtm, bbtm, fWhbm, bhbm,
      fW0, W0, b0, fW1, b1, out_bonds);
}
